// NearestNeighbours_65360812311239
// MI455X (gfx1250) — matrix operand order repaired and checked by reading, not compiled, not run on hardware
//
#include <hip/hip_runtime.h>
#include <math.h>
#include <stdint.h>

// ---------------------------------------------------------------------------
// Problem constants (from the reference)
// ---------------------------------------------------------------------------
#define N_USERS 50000
#define N_ITEMS 4000
#define BATCH   256
#define NK      10          // num_neighbors (reference passes 10)
#define EPSV    1e-8f

// GEMM tiling: K-chunk staged in LDS. 4000 = 25 * 160, 160 = 5 * 32 (wmma K)
#define KC      160
#define NIT     (N_ITEMS / KC)   // 25 K-chunk iterations
#define GTHREADS 256             // GEMM block: 8 waves, 2 M-tiles per wave

// ---------------------------------------------------------------------------
// Workspace layout (bytes, 256B aligned)
// ---------------------------------------------------------------------------
static constexpr size_t OFF_RNORMA = 0;          // 50000 f32 (reciprocal norms)
static constexpr size_t OFF_RNORMU = 204800;     // 256 f32   (reciprocal norms)
static constexpr size_t OFF_UBF    = 208896;     // 256*4000 bf16
static constexpr size_t OFF_SIM    = 2260992;    // 256*50000 f32
static constexpr size_t OFF_TOPW   = 53460992;   // 256*10 f32
static constexpr size_t OFF_TOPI   = 53471232;   // 256*10 i32

// ---------------------------------------------------------------------------
// Vector types for WMMA / TDM
// ---------------------------------------------------------------------------
typedef __bf16 v16bf __attribute__((ext_vector_type(16)));
typedef __bf16 v8bf  __attribute__((ext_vector_type(8)));
typedef __bf16 v2bf  __attribute__((ext_vector_type(2)));
typedef float  v8f   __attribute__((ext_vector_type(8)));
typedef unsigned int u32x4 __attribute__((ext_vector_type(4)));
typedef int          i32x4 __attribute__((ext_vector_type(4)));
typedef int          i32x8 __attribute__((ext_vector_type(8)));

// ---------------------------------------------------------------------------
// K1: reciprocal row sum-of-squares. rnorma[u] = 1 / max(sum(F[u,:]^2), eps)
// One wave per row, 8 waves per block. 50000 = 6250 * 8.
// ---------------------------------------------------------------------------
__global__ __launch_bounds__(256) void k_row_norms(const float* __restrict__ F,
                                                   float* __restrict__ rnorma) {
  const int wave = threadIdx.x >> 5;
  const int lane = threadIdx.x & 31;
  const int r    = blockIdx.x * 8 + wave;
  const float* row = F + (size_t)r * N_ITEMS;
  float s = 0.0f;
  for (int k = lane * 4; k < N_ITEMS; k += 32 * 4) {
    float4 v = *reinterpret_cast<const float4*>(row + k);
    s += v.x * v.x + v.y * v.y + v.z * v.z + v.w * v.w;
  }
#pragma unroll
  for (int m = 16; m > 0; m >>= 1) s += __shfl_xor(s, m, 32);
  if (lane == 0) rnorma[r] = 1.0f / fmaxf(s, EPSV);
}

// ---------------------------------------------------------------------------
// K2: gather selected user rows -> bf16; rnormu[b] = rnorma[uid[b]]
// ---------------------------------------------------------------------------
__global__ __launch_bounds__(256) void k_gather_users(const float* __restrict__ F,
                                                      const int* __restrict__ uid,
                                                      const float* __restrict__ rnorma,
                                                      __bf16* __restrict__ ubf,
                                                      float* __restrict__ rnormu) {
  const int b = blockIdx.x;
  const int u = uid[b];
  const float* row = F + (size_t)u * N_ITEMS;
  for (int k = threadIdx.x * 2; k < N_ITEMS; k += 512) {
    float2 v = *reinterpret_cast<const float2*>(row + k);
    v2bf p;
    p[0] = (__bf16)v.x;
    p[1] = (__bf16)v.y;
    *reinterpret_cast<v2bf*>(&ubf[(size_t)b * N_ITEMS + k]) = p;
  }
  if (threadIdx.x == 0) rnormu[b] = rnorma[u];
}

// ---------------------------------------------------------------------------
// TDM descriptor issue: 2D tile, 16 rows x KC cols of fp32 from F into LDS.
// ---------------------------------------------------------------------------
#if __has_builtin(__builtin_amdgcn_tensor_load_to_lds)
#define HAVE_TDM 1
__device__ __forceinline__ void tdm_load_tile(const float* F, int n0, int kbase,
                                              unsigned ldsa) {
  unsigned long long ga =
      (unsigned long long)(uintptr_t)(F + (size_t)n0 * N_ITEMS + kbase);
  u32x4 g0 = {1u,                                  // count=1, user descriptor
              ldsa,                                // lds_addr
              (unsigned)ga,                        // global_addr[31:0]
              (unsigned)(ga >> 32) | (2u << 30)};  // ga[56:32] | type=2
  i32x8 g1 = {(int)(2u << 16),                     // data_size=4B
              (int)(((unsigned)N_ITEMS & 0xFFFFu) << 16),               // dim0 lo
              (int)(((unsigned)N_ITEMS >> 16) |
                    (((unsigned)N_USERS & 0xFFFFu) << 16)),             // dim0 hi | dim1 lo
              (int)(((unsigned)N_USERS >> 16) | ((unsigned)KC << 16)),  // dim1 hi | tile0
              16,                                  // tile_dim1 = 16 rows
              N_ITEMS,                             // dim0 stride
              0, 0};
  i32x4 g2 = {0, 0, 0, 0};
  i32x4 g3 = {0, 0, 0, 0};
#if __clang_major__ >= 23
  i32x8 g4 = {0, 0, 0, 0, 0, 0, 0, 0};
  __builtin_amdgcn_tensor_load_to_lds(g0, g1, g2, g3, g4, 0);
#else
  __builtin_amdgcn_tensor_load_to_lds(g0, g1, g2, g3, 0);
#endif
}
#else
#define HAVE_TDM 0
#endif

// ---------------------------------------------------------------------------
// K3: sim[m,n] = dot(users[m,:], F[n,:]) * rnormu[m] * rnorma[n]
// Grid: 3125 blocks x 256 threads (8 waves). Block owns 16 F rows (N-tile);
// wave w computes TWO 16x16 C tiles (M = 32w and 32w+16), sharing each
// B fragment between two WMMAs (halves LDS reads, lengthens acc dep chains).
// Pipeline per K-chunk: TDM fp32 tile -> LDS (double buffered) ||
// cooperative fp32->bf16 convert into shared bf16 tile -> WMMA fragments.
// ---------------------------------------------------------------------------
__global__ __launch_bounds__(GTHREADS) void k_gemm_sim(
    const float* __restrict__ F, const __bf16* __restrict__ ubf,
    const float* __restrict__ rnormu, const float* __restrict__ rnorma,
    float* __restrict__ sim) {
  const int n0   = blockIdx.x * 16;
  const int tid  = threadIdx.x;
  const int wave = tid >> 5;
  const int lane = tid & 31;
  const int m0   = wave * 32;          // this wave's first M tile

#if HAVE_TDM
  __shared__ __align__(16) float fbuf[2][16 * KC];  // TDM destinations (fp32)
#endif
  __shared__ __align__(16) __bf16 bbuf[16 * KC];    // converted bf16 tile

  v8f acc0 = {0.f, 0.f, 0.f, 0.f, 0.f, 0.f, 0.f, 0.f};
  v8f acc1 = {0.f, 0.f, 0.f, 0.f, 0.f, 0.f, 0.f, 0.f};

  // A-matrix 16x32 bf16 lane layout (ISA 7.12.2):
  //  lane<16 : M=lane,     elems 0-7 K=0..7,  elems 8-15 K=16..23
  //  lane>=16: M=lane-16,  elems 0-7 K=8..15, elems 8-15 K=24..31
  const int mrow0 = m0 + (lane & 15);
  const int mrow1 = mrow0 + 16;
  const int koff  = (lane >> 4) * 8;
  // B-matrix 32x16 bf16 lane layout (same K order as A): lanes 0-15 K=0..7,16..23
  // for N=lane; lanes 16-31 K=8..15,24..31 for N=lane-16.
  const int nrow  = lane & 15;
  const int khalf = (lane >> 4) * 8;

#if HAVE_TDM
  if (wave == 0) tdm_load_tile(F, n0, 0, (unsigned)(uintptr_t)&fbuf[0][0]);
#endif

  for (int it = 0; it < NIT; ++it) {
    const int kbase = it * KC;

#if HAVE_TDM
    if (wave == 0) __builtin_amdgcn_s_wait_tensorcnt(0);   // tile it landed
#endif
    __syncthreads();   // tile it ready; bbuf + other fbuf free (prev compute done)

#if HAVE_TDM
    if (wave == 0 && it + 1 < NIT)
      tdm_load_tile(F, n0, kbase + KC,
                    (unsigned)(uintptr_t)&fbuf[(it + 1) & 1][0]);
    // Cooperative fp32 -> bf16 conversion of tile it (once per workgroup).
    {
      const float* fb = &fbuf[it & 1][0];
      for (int i = tid; i < (16 * KC) / 2; i += GTHREADS) {
        float2 v = *reinterpret_cast<const float2*>(fb + 2 * i);
        v2bf p;
        p[0] = (__bf16)v.x;
        p[1] = (__bf16)v.y;
        *reinterpret_cast<v2bf*>(&bbuf[2 * i]) = p;
      }
    }
#else
    // Fallback: cooperative global -> bf16 LDS copy/convert.
    for (int i = tid; i < (16 * KC) / 2; i += GTHREADS) {
      const int r = (2 * i) / KC, c = (2 * i) % KC;   // KC even: pair in one row
      float2 v = *reinterpret_cast<const float2*>(
          &F[(size_t)(n0 + r) * N_ITEMS + kbase + c]);
      v2bf p;
      p[0] = (__bf16)v.x;
      p[1] = (__bf16)v.y;
      *reinterpret_cast<v2bf*>(&bbuf[2 * i]) = p;
    }
#endif
    __syncthreads();   // bf16 tile ready for all waves

#pragma unroll
    for (int s = 0; s < KC / 32; ++s) {
      const int ka = kbase + s * 32 + koff;
      // ---- A fragments for both M tiles (16B bf16 loads, L2-resident)
      const __bf16* ap0 = ubf + (size_t)mrow0 * N_ITEMS + ka;
      v8bf a0lo = *reinterpret_cast<const v8bf*>(ap0);
      v8bf a0hi = *reinterpret_cast<const v8bf*>(ap0 + 16);
      v16bf a0 = __builtin_shufflevector(a0lo, a0hi, 0, 1, 2, 3, 4, 5, 6, 7,
                                         8, 9, 10, 11, 12, 13, 14, 15);
      const __bf16* ap1 = ubf + (size_t)mrow1 * N_ITEMS + ka;
      v8bf a1lo = *reinterpret_cast<const v8bf*>(ap1);
      v8bf a1hi = *reinterpret_cast<const v8bf*>(ap1 + 16);
      v16bf a1 = __builtin_shufflevector(a1lo, a1hi, 0, 1, 2, 3, 4, 5, 6, 7,
                                         8, 9, 10, 11, 12, 13, 14, 15);
      // ---- shared B fragment: 8 bf16 at khalf and 8 at khalf+16 from the shared tile
      const __bf16* bp = &bbuf[nrow * KC + s * 32 + khalf];
      v8bf blo = *reinterpret_cast<const v8bf*>(bp);
      v8bf bhi = *reinterpret_cast<const v8bf*>(bp + 16);
      v16bf b = __builtin_shufflevector(blo, bhi, 0, 1, 2, 3, 4, 5, 6, 7,
                                        8, 9, 10, 11, 12, 13, 14, 15);

      acc0 = __builtin_amdgcn_wmma_f32_16x16x32_bf16(
          false, a0, false, b, (short)0, acc0, false, false);
      acc1 = __builtin_amdgcn_wmma_f32_16x16x32_bf16(
          false, a1, false, b, (short)0, acc1, false, false);
    }
  }

  // Epilogue: C layout — VGPR j: lanes0-15 M=j, lanes16-31 M=j+8; N=lane&15
  // Pure multiplies: reciprocal norms precomputed.
  const float rna = rnorma[n0 + nrow];
  const int mb = (lane >> 4) * 8;
#pragma unroll
  for (int j = 0; j < 8; ++j) {
    const int m = m0 + mb + j;
    sim[(size_t)m * N_USERS + (n0 + nrow)] = acc0[j] * (rnormu[m] * rna);
  }
#pragma unroll
  for (int j = 0; j < 8; ++j) {
    const int m = m0 + 16 + mb + j;
    sim[(size_t)m * N_USERS + (n0 + nrow)] = acc1[j] * (rnormu[m] * rna);
  }
}

// ---------------------------------------------------------------------------
// K4: per batch row, top-10 of sim[b,:] with JAX tie rule (lower index wins),
// then normalized weights. One block of 256 threads per row.
// Per-thread top-10 kept in registers via fully unrolled insertion.
// ---------------------------------------------------------------------------
__global__ __launch_bounds__(256) void k_topk(const float* __restrict__ sim,
                                              float* __restrict__ topw,
                                              int* __restrict__ topi) {
  const int b = blockIdx.x;
  const int t = threadIdx.x;
  __shared__ float cv[256 * NK];
  __shared__ int   ci[256 * NK];
  __shared__ float rv[256];
  __shared__ int   ri[256];
  __shared__ int   rs[256];
  __shared__ float wv[NK];
  __shared__ int   wi[NK];

  float lv[NK];
  int   li[NK];
#pragma unroll
  for (int j = 0; j < NK; ++j) { lv[j] = -__builtin_inff(); li[j] = 0x7fffffff; }

  const float* srow = sim + (size_t)b * N_USERS;
  for (int u = t; u < N_USERS; u += 256) {
    const float s = srow[u];
    if (s > lv[NK - 1]) {
      float cur = s; int curi = u;
#pragma unroll
      for (int j = 0; j < NK; ++j) {        // sorted-desc insertion, strict >
        const float tv = lv[j]; const int ti = li[j];
        const bool ins = (cur > tv);
        lv[j] = ins ? cur : tv;
        li[j] = ins ? curi : ti;
        cur   = ins ? tv : cur;
        curi  = ins ? ti : curi;
      }
    }
  }
#pragma unroll
  for (int j = 0; j < NK; ++j) { cv[t * NK + j] = lv[j]; ci[t * NK + j] = li[j]; }
  __syncthreads();

  for (int k = 0; k < NK; ++k) {
    float bv = -__builtin_inff(); int bi = 0x7fffffff; int bs = t * NK;
#pragma unroll
    for (int j = 0; j < NK; ++j) {
      const float v = cv[t * NK + j];
      const int  id = ci[t * NK + j];
      if (v > bv || (v == bv && id < bi)) { bv = v; bi = id; bs = t * NK + j; }
    }
    rv[t] = bv; ri[t] = bi; rs[t] = bs;
    __syncthreads();
    for (int s = 128; s > 0; s >>= 1) {
      if (t < s) {
        if (rv[t + s] > rv[t] || (rv[t + s] == rv[t] && ri[t + s] < ri[t])) {
          rv[t] = rv[t + s]; ri[t] = ri[t + s]; rs[t] = rs[t + s];
        }
      }
      __syncthreads();
    }
    if (t == 0) { wv[k] = rv[0]; wi[k] = ri[0]; cv[rs[0]] = -__builtin_inff(); }
    __syncthreads();
  }

  if (t == 0) {
    float sum = 0.0f;
    for (int k = 0; k < NK; ++k) sum += wv[k];
    for (int k = 0; k < NK; ++k) {
      topw[b * NK + k] = wv[k] / sum;
      topi[b * NK + k] = wi[k];
    }
  }
}

// ---------------------------------------------------------------------------
// K5: ratings[b,i] = sum_k w[b,k] * F[idx[b,k], i]
// ---------------------------------------------------------------------------
__global__ __launch_bounds__(256) void k_ratings(const float* __restrict__ F,
                                                 const float* __restrict__ topw,
                                                 const int* __restrict__ topi,
                                                 float* __restrict__ out) {
  const int b = blockIdx.x;
  __shared__ float w[NK];
  __shared__ int   id[NK];
  if (threadIdx.x < NK) {
    w[threadIdx.x]  = topw[b * NK + threadIdx.x];
    id[threadIdx.x] = topi[b * NK + threadIdx.x];
  }
  __syncthreads();
  for (int i = threadIdx.x; i < N_ITEMS; i += 256) {
    float acc = 0.0f;
#pragma unroll
    for (int k = 0; k < NK; ++k) acc += w[k] * F[(size_t)id[k] * N_ITEMS + i];
    out[(size_t)b * N_ITEMS + i] = acc;
  }
}

// ---------------------------------------------------------------------------
// Launcher
// ---------------------------------------------------------------------------
extern "C" void kernel_launch(void* const* d_in, const int* in_sizes, int n_in,
                              void* d_out, int out_size, void* d_ws, size_t ws_size,
                              hipStream_t stream) {
  (void)in_sizes; (void)n_in; (void)out_size; (void)ws_size;
  const float* F   = (const float*)d_in[0];
  const int*   uid = (const int*)d_in[1];
  // d_in[2] = num_neighbors (==10 per reference); NK is compile-time.

  char* ws = (char*)d_ws;
  float*  rnorma = (float*)(ws + OFF_RNORMA);
  float*  rnormu = (float*)(ws + OFF_RNORMU);
  __bf16* ubf    = (__bf16*)(ws + OFF_UBF);
  float*  sim    = (float*)(ws + OFF_SIM);
  float*  topw   = (float*)(ws + OFF_TOPW);
  int*    topi   = (int*)(ws + OFF_TOPI);
  float*  out    = (float*)d_out;

  k_row_norms<<<N_USERS / 8, 256, 0, stream>>>(F, rnorma);
  k_gather_users<<<BATCH, 256, 0, stream>>>(F, uid, rnorma, ubf, rnormu);
  k_gemm_sim<<<N_USERS / 16, GTHREADS, 0, stream>>>(F, ubf, rnormu, rnorma, sim);
  k_topk<<<BATCH, 256, 0, stream>>>(sim, topw, topi);
  k_ratings<<<BATCH, 256, 0, stream>>>(F, topw, topi, out);
}
